// SparseExpertBank_4501125726437
// MI455X (gfx1250) — matrix operand order repaired and checked by reading, not compiled, not run on hardware
//
#include <hip/hip_runtime.h>

// ---- plain ext-vector types (trivially copyable, union-safe) ----
typedef __attribute__((ext_vector_type(16))) __bf16        v16bf;
typedef __attribute__((ext_vector_type(8)))  float         v8f;
typedef __attribute__((ext_vector_type(4)))  float         v4f;
typedef __attribute__((ext_vector_type(4)))  unsigned int  v4u;
typedef __attribute__((ext_vector_type(2)))  unsigned int  v2u;

// GCC-style vector to match the async-LDS builtin's parameter type exactly
typedef int v4i_raw __attribute__((__vector_size__(16)));

#define AS1 __attribute__((address_space(1)))
#define AS3 __attribute__((address_space(3)))

#if defined(__AMDGCN__) && __has_builtin(__builtin_amdgcn_global_load_async_to_lds_b128)
#define USE_ASYNC_COPY 1
#else
#define USE_ASYNC_COPY 0
#endif

static constexpr int E  = 8;
static constexpr int C  = 2048;
static constexpr int DM = 1024;   // d_model
static constexpr int DF = 4096;   // d_ff
static constexpr int LDK = 40;    // LDS row stride in u16 (32 data + 8 pad -> 80B, 16B aligned)

// fp32 -> bf16 (round to nearest even)
__device__ __forceinline__ unsigned short f2bf(float f) {
  unsigned int x = __builtin_bit_cast(unsigned int, f);
  x = x + 0x7FFFu + ((x >> 16) & 1u);
  return (unsigned short)(x >> 16);
}
__device__ __forceinline__ unsigned int pack_bf16x2(float lo, float hi) {
  unsigned int xa = __builtin_bit_cast(unsigned int, lo);
  unsigned int xb = __builtin_bit_cast(unsigned int, hi);
  xa = xa + 0x7FFFu + ((xa >> 16) & 1u);
  xb = xb + 0x7FFFu + ((xb >> 16) & 1u);
  return (xa >> 16) | (xb & 0xFFFF0000u);
}

// A-fragment (16x32 bf16, M x K): lane m<16 holds K {0..7,16..23}, lane m+16 holds {8..15,24..31}
__device__ __forceinline__ v16bf load_frag_a(const unsigned short* s, int row, int laneHalf) {
  const v4u* p = reinterpret_cast<const v4u*>(s + row * LDK + laneHalf * 8);
  union { v4u q[2]; v16bf v; } u;
  u.q[0] = p[0];   // bytes [laneHalf?16:0 .. +16)
  u.q[1] = p[2];   // +32 bytes
  return u.v;
}
// B-fragment (32x16 bf16, K x N): lane n<16 holds column n K {0..7,16..23}, lane n+16 holds {8..15,24..31}
__device__ __forceinline__ v16bf load_frag_b(const unsigned short* s, int row, int laneHalf) {
  const v4u* p = reinterpret_cast<const v4u*>(s + row * LDK + laneHalf * 8);
  union { v4u q[2]; v16bf v; } u;
  u.q[0] = p[0];
  u.q[1] = p[2];
  return u.v;
}

__device__ __forceinline__ v8f wmma_bf16(v16bf a, v16bf b, v8f c) {
  return __builtin_amdgcn_wmma_f32_16x16x32_bf16(false, a, false, b, (short)0, c, false, false);
}

// ---------------- Kernel 1: hidden = silu(X*Wg^T) * (X*Wu^T)  (bf16 out) ----------------
__global__ __launch_bounds__(256)
void moe_gateup_kernel(const float* __restrict__ X, const float* __restrict__ Wg,
                       const float* __restrict__ Wu, unsigned short* __restrict__ H) {
  __shared__ unsigned short sA[128 * LDK];
  __shared__ unsigned short sG[128 * LDK];
  __shared__ unsigned short sU[128 * LDK];

  const int e   = blockIdx.z;
  const int cm0 = blockIdx.y * 128;
  const int fn0 = blockIdx.x * 128;

  const float* Xe  = X  + ((size_t)e * C  + cm0) * DM;
  const float* Wge = Wg + ((size_t)e * DF + fn0) * DM;
  const float* Wue = Wu + ((size_t)e * DF + fn0) * DM;
  unsigned short* Ho = H + ((size_t)e * C + cm0) * DF + fn0;

  const int tid = threadIdx.x;
  const int lane = tid & 31;
  const int w = tid >> 5;
  const int wm = (w & 1) * 64;      // wave M offset
  const int wn = (w >> 1) * 32;     // wave N offset
  const int laneHalf = lane >> 4;
  const int laneRow  = lane & 15;

  v8f accG[4][2], accU[4][2];
  v8f zero = {};
#pragma unroll
  for (int mi = 0; mi < 4; ++mi)
#pragma unroll
    for (int ni = 0; ni < 2; ++ni) { accG[mi][ni] = zero; accU[mi][ni] = zero; }

  for (int k0 = 0; k0 < DM; k0 += 32) {
    if (k0 + 32 < DM) {   // prefetch next K-slab of the weight streams
      const size_t prow = (size_t)(tid >> 1) * DM + (size_t)(k0 + 32);
      __builtin_prefetch(Wge + prow, 0, 0);
      __builtin_prefetch(Wue + prow, 0, 0);
    }
#pragma unroll
    for (int i = 0; i < 4; ++i) {
      const int idx = tid + i * 256;          // 0..1023
      const int row = idx >> 3;               // 0..127
      const int c4  = idx & 7;                // float4 index within 32-wide slab
      const size_t go = (size_t)row * DM + k0 + c4 * 4;
      v4f fa = *reinterpret_cast<const v4f*>(Xe  + go);
      v4f fg = *reinterpret_cast<const v4f*>(Wge + go);
      v4f fu = *reinterpret_cast<const v4f*>(Wue + go);
      v2u pa; pa.x = pack_bf16x2(fa.x, fa.y); pa.y = pack_bf16x2(fa.z, fa.w);
      v2u pg; pg.x = pack_bf16x2(fg.x, fg.y); pg.y = pack_bf16x2(fg.z, fg.w);
      v2u pu; pu.x = pack_bf16x2(fu.x, fu.y); pu.y = pack_bf16x2(fu.z, fu.w);
      const int so = row * LDK + c4 * 4;
      *reinterpret_cast<v2u*>(sA + so) = pa;
      *reinterpret_cast<v2u*>(sG + so) = pg;
      *reinterpret_cast<v2u*>(sU + so) = pu;
    }
    __syncthreads();

    v16bf bG[2], bU[2];
#pragma unroll
    for (int ni = 0; ni < 2; ++ni) {
      bG[ni] = load_frag_b(sG, wn + ni * 16 + laneRow, laneHalf);
      bU[ni] = load_frag_b(sU, wn + ni * 16 + laneRow, laneHalf);
    }
#pragma unroll
    for (int mi = 0; mi < 4; ++mi) {
      v16bf a = load_frag_a(sA, wm + mi * 16 + laneRow, laneHalf);
#pragma unroll
      for (int ni = 0; ni < 2; ++ni) {
        accG[mi][ni] = wmma_bf16(a, bG[ni], accG[mi][ni]);
        accU[mi][ni] = wmma_bf16(a, bU[ni], accU[mi][ni]);
      }
    }
    __syncthreads();
  }

  // epilogue: silu(gate) * up -> bf16
#pragma unroll
  for (int mi = 0; mi < 4; ++mi)
#pragma unroll
    for (int ni = 0; ni < 2; ++ni)
#pragma unroll
      for (int j = 0; j < 8; ++j) {
        const float g  = accG[mi][ni][j];
        const float uu = accU[mi][ni][j];
        const float h  = g * (1.0f / (1.0f + __expf(-g))) * uu;
        const int row = wm + mi * 16 + laneHalf * 8 + j;
        const int col = wn + ni * 16 + laneRow;
        Ho[(size_t)row * DF + col] = f2bf(h);
      }
}

// ---------------- Kernel 2: out = hidden * Wd^T  (fp32 out) ----------------
__global__ __launch_bounds__(256)
void moe_down_kernel(const unsigned short* __restrict__ H, const float* __restrict__ Wd,
                     float* __restrict__ Out) {
  __shared__ unsigned short sA[128 * LDK];
  __shared__ unsigned short sB[128 * LDK];

  const int e   = blockIdx.z;
  const int cm0 = blockIdx.y * 128;
  const int dn0 = blockIdx.x * 128;

  const unsigned short* He = H + ((size_t)e * C + cm0) * DF;
  const float* Wde = Wd + ((size_t)e * DM + dn0) * DF;
  float* Oo = Out + ((size_t)e * C + cm0) * DM + dn0;

  const int tid = threadIdx.x;
  const int lane = tid & 31;
  const int w = tid >> 5;
  const int wm = (w & 1) * 64;
  const int wn = (w >> 1) * 32;
  const int laneHalf = lane >> 4;
  const int laneRow  = lane & 15;

  v8f acc[4][2];
  v8f zero = {};
#pragma unroll
  for (int mi = 0; mi < 4; ++mi)
#pragma unroll
    for (int ni = 0; ni < 2; ++ni) acc[mi][ni] = zero;

  for (int k0 = 0; k0 < DF; k0 += 32) {
    if (k0 + 32 < DF) {
      __builtin_prefetch(Wde + (size_t)(tid >> 1) * DF + (size_t)(k0 + 32), 0, 0);
    }

    // ---- A tile (bf16 hidden): straight copy -> async global->LDS DMA if available ----
#if USE_ASYNC_COPY
#pragma unroll
    for (int i = 0; i < 2; ++i) {
      const int idx = tid + i * 256;          // 0..511
      const int row = idx >> 2;               // 0..127
      const int c   = idx & 3;                // 16B chunk within 64B row slab
      const unsigned short* gsrc = He + (size_t)row * DF + k0 + c * 8;
      unsigned short*       ldst = sA + row * LDK + c * 8;
      __builtin_amdgcn_global_load_async_to_lds_b128(
          (AS1 v4i_raw*)(const AS1 void*)gsrc,
          (AS3 v4i_raw*)(AS3 void*)ldst,
          0, 0);
    }
#else
#pragma unroll
    for (int i = 0; i < 4; ++i) {
      const int idx = tid + i * 256;
      const int row = idx >> 3;
      const int c4  = idx & 7;
      v2u ha = *reinterpret_cast<const v2u*>(He + (size_t)row * DF + k0 + c4 * 4);
      *reinterpret_cast<v2u*>(sA + row * LDK + c4 * 4) = ha;
    }
#endif

    // ---- B tile (fp32 w_down): convert fp32 -> bf16 on the fly ----
#pragma unroll
    for (int i = 0; i < 4; ++i) {
      const int idx = tid + i * 256;
      const int row = idx >> 3;
      const int c4  = idx & 7;
      v4f fw = *reinterpret_cast<const v4f*>(Wde + (size_t)row * DF + k0 + c4 * 4);
      v2u pb; pb.x = pack_bf16x2(fw.x, fw.y); pb.y = pack_bf16x2(fw.z, fw.w);
      *reinterpret_cast<v2u*>(sB + row * LDK + c4 * 4) = pb;
    }

#if USE_ASYNC_COPY
#if __has_builtin(__builtin_amdgcn_s_wait_asynccnt)
    __builtin_amdgcn_s_wait_asynccnt(0);
#else
    asm volatile("s_wait_asynccnt 0" ::: "memory");
#endif
#endif
    __syncthreads();

    v16bf bF[2];
#pragma unroll
    for (int ni = 0; ni < 2; ++ni)
      bF[ni] = load_frag_b(sB, wn + ni * 16 + laneRow, laneHalf);
#pragma unroll
    for (int mi = 0; mi < 4; ++mi) {
      v16bf a = load_frag_a(sA, wm + mi * 16 + laneRow, laneHalf);
#pragma unroll
      for (int ni = 0; ni < 2; ++ni)
        acc[mi][ni] = wmma_bf16(a, bF[ni], acc[mi][ni]);
    }
    __syncthreads();
  }

#pragma unroll
  for (int mi = 0; mi < 4; ++mi)
#pragma unroll
    for (int ni = 0; ni < 2; ++ni)
#pragma unroll
      for (int j = 0; j < 8; ++j) {
        const int row = wm + mi * 16 + laneHalf * 8 + j;
        const int col = wn + ni * 16 + laneRow;
        Oo[(size_t)row * DM + col] = acc[mi][ni][j];
      }
}

extern "C" void kernel_launch(void* const* d_in, const int* in_sizes, int n_in,
                              void* d_out, int out_size, void* d_ws, size_t ws_size,
                              hipStream_t stream) {
  (void)in_sizes; (void)n_in; (void)out_size; (void)ws_size;
  const float* X  = (const float*)d_in[0];   // (E, C, DM)
  const float* Wg = (const float*)d_in[1];   // (E, DF, DM)
  const float* Wu = (const float*)d_in[2];   // (E, DF, DM)
  const float* Wd = (const float*)d_in[3];   // (E, DM, DF)
  float* Out = (float*)d_out;                // (E, C, DM)
  unsigned short* Hws = (unsigned short*)d_ws;  // (E, C, DF) bf16, 128 MB

  dim3 blk(256);
  dim3 gA(DF / 128, C / 128, E);   // 32 x 16 x 8
  dim3 gB(DM / 128, C / 128, E);   //  8 x 16 x 8
  hipLaunchKernelGGL(moe_gateup_kernel, gA, blk, 0, stream, X, Wg, Wu, Hws);
  hipLaunchKernelGGL(moe_down_kernel,   gB, blk, 0, stream, Hws, Wd, Out);
}
